// MODEL_7301444403692
// MI455X (gfx1250) — matrix operand order repaired and checked by reading, not compiled, not run on hardware
//
#include <hip/hip_runtime.h>
#include <stdint.h>

typedef __attribute__((ext_vector_type(16))) __bf16 v16bf;
typedef __attribute__((ext_vector_type(8)))  float  v8f;

#define BSZ 2048
#define LQ  100
#define LP2 128    // padded L (K dim of GEMM3)
#define DIM 256
#define KC  32
#define NC  16

// LDS strides (elements)
#define HS  264    // hist bf16
#define PS  264    // proj bf16
#define IS  264    // interests bf16
#define WTS 120    // weights f32 (per k row over l)
#define WBS 136    // weights bf16 (padded to 128 cols)
#define AWS 40     // aw rows

// LDS byte offsets (all 16B aligned)
#define OFF_HIST 0                         // 128*264*2 = 67584
#define OFF_WTB  67584                     // 32*136*2  = 8704
#define OFF_WT   76288                     // 32*120*4  = 15360
#define OFF_PROJ 91648                     // 112*264*2 = 59136 (dead after GEMM2)
#define OFF_INT  91648                     // 32*264*2  = 16896 (union with proj)
#define OFF_AW   108544                    // 16*40*4   = 2560
#define OFF_AWB  111104                    // 16*40*2   = 1280
#define SMEM_BYTES 150784

__device__ __forceinline__ uint16_t f2bf_bits(float x) {
  uint32_t u = __float_as_uint(x);
  uint32_t r = (u + 0x7FFFu + ((u >> 16) & 1u)) >> 16;  // round-to-nearest-even
  return (uint16_t)r;
}

// branch-free tanh: one v_exp_f32 + fast divide; exact ±1 saturation via exp underflow
__device__ __forceinline__ float fast_tanh(float x) {
  float ax = fabsf(x);
  float e  = __expf(-2.0f * ax);
  float r  = __fdividef(1.0f - e, 1.0f + e);
  return copysignf(r, x);
}

__device__ __forceinline__ v8f wmma_bf16(v16bf a, v16bf b, v8f c) {
  return __builtin_amdgcn_wmma_f32_16x16x32_bf16(false, a, false, b, (short)0, c, false, false);
}

// A 16x32 bf16 fragment from LDS. lanes 0-15: M=lane, K 0-15; lanes 16-31: M=lane-16, K 16-31
__device__ __forceinline__ v16bf load_a_lds(const uint16_t* p, int stride, int rb, int kb, int lane) {
  int m = lane & 15, h = lane >> 4;
  const uint16_t* row = p + (rb + m) * stride + kb + h * 16;
  union { v16bf v; uint4 q[2]; } f;
  f.q[0] = *(const uint4*)(row);
  f.q[1] = *(const uint4*)(row + 8);
  return f.v;
}

// A fragment from global f32 row pointer already offset by (M row, + h*16)
__device__ __forceinline__ v16bf load_a_g_f32(const float* p) {
  union { v16bf v; uint16_t s[16]; } f;
#pragma unroll
  for (int i = 0; i < 16; i += 4) {
    float4 x = *(const float4*)(p + i);
    f.s[i] = f2bf_bits(x.x); f.s[i+1] = f2bf_bits(x.y); f.s[i+2] = f2bf_bits(x.z); f.s[i+3] = f2bf_bits(x.w);
  }
  return f.v;
}

// B 32x16 bf16 fragment: per-lane 16 consecutive K values (lane = column, lane half = K half).
// Source has K contiguous in memory: 16 consecutive f32 from global.
__device__ __forceinline__ v16bf load_b16_g_f32(const float* p) {
  union { v16bf v; uint16_t s[16]; } f;
#pragma unroll
  for (int i = 0; i < 16; i += 4) {
    float4 x = *(const float4*)(p + i);
    f.s[i] = f2bf_bits(x.x); f.s[i+1] = f2bf_bits(x.y); f.s[i+2] = f2bf_bits(x.z); f.s[i+3] = f2bf_bits(x.w);
  }
  return f.v;
}

// B fragment: 16 consecutive bf16 from LDS (K contiguous)
__device__ __forceinline__ v16bf load_b16_lds(const uint16_t* p) {
  union { v16bf v; uint4 q[2]; } f;
  f.q[0] = *(const uint4*)(p);
  f.q[1] = *(const uint4*)(p + 8);
  return f.v;
}

// B fragment: 16 K values strided in LDS (K = row index of a row-major array)
__device__ __forceinline__ v16bf load_b16_lds_strided(const uint16_t* p, int stride) {
  union { v16bf v; uint16_t s[16]; } f;
#pragma unroll
  for (int i = 0; i < 16; ++i) f.s[i] = p[i * stride];
  return f.v;
}

__global__ __launch_bounds__(256)
void poly_attn_kernel(const float* __restrict__ hist,
                      const float* __restrict__ cand,
                      const int*   __restrict__ num_int,
                      const int*   __restrict__ cat_cnt,
                      const float* __restrict__ Wp,
                      const float* __restrict__ cc,
                      float*       __restrict__ out) {
  extern __shared__ __align__(16) char smem[];
  uint16_t* sh_hist = (uint16_t*)(smem + OFF_HIST);
  uint16_t* sh_wtb  = (uint16_t*)(smem + OFF_WTB);
  float*    sh_wt   = (float*)   (smem + OFF_WT);
  uint16_t* sh_proj = (uint16_t*)(smem + OFF_PROJ);
  uint16_t* sh_int  = (uint16_t*)(smem + OFF_INT);
  float*    sh_aw   = (float*)   (smem + OFF_AW);
  uint16_t* sh_awb  = (uint16_t*)(smem + OFF_AWB);

  const int b    = blockIdx.x;
  const int tid  = threadIdx.x;
  const int lane = tid & 31;
  const int w    = tid >> 5;
  const int m15  = lane & 15;
  const int hi   = lane >> 4;

  // ---- Step 0: hist -> bf16 LDS, rows [100,128) zeroed ----
  {
    const float* hb = hist + (size_t)b * LQ * DIM;
    for (int idx = tid; idx < 128 * 64; idx += 256) {
      int r = idx >> 6, c4 = (idx & 63) << 2;
      float4 x = (r < LQ) ? *(const float4*)(hb + r * DIM + c4) : make_float4(0.f, 0.f, 0.f, 0.f);
      uint2 pk;
      pk.x = (uint32_t)f2bf_bits(x.x) | ((uint32_t)f2bf_bits(x.y) << 16);
      pk.y = (uint32_t)f2bf_bits(x.z) | ((uint32_t)f2bf_bits(x.w) << 16);
      *(uint2*)(sh_hist + r * HS + c4) = pk;
    }
  }
  __syncthreads();

  // ---- Step 1: GEMM1 proj[l][e] = tanh(sum_d hist[l][d] * W[e][d]) ----
  for (int c = w; c < 16; c += 8) {
    v16bf bfrag[8];
    {
      const float* wrow = Wp + (size_t)(c * 16 + m15) * DIM + hi * 16;
#pragma unroll
      for (int kk = 0; kk < 8; ++kk) bfrag[kk] = load_b16_g_f32(wrow + kk * 32);
    }
    for (int r = 0; r < 7; ++r) {
      v16bf afr[8];
#pragma unroll
      for (int kk = 0; kk < 8; ++kk)
        afr[kk] = load_a_lds(sh_hist, HS, r * 16, kk * 32, lane);
      v8f acc0 = {0.f, 0.f, 0.f, 0.f, 0.f, 0.f, 0.f, 0.f};
      v8f acc1 = {0.f, 0.f, 0.f, 0.f, 0.f, 0.f, 0.f, 0.f};
#pragma unroll
      for (int kk = 0; kk < 8; kk += 2) {
        acc0 = wmma_bf16(afr[kk],     bfrag[kk],     acc0);
        acc1 = wmma_bf16(afr[kk + 1], bfrag[kk + 1], acc1);
      }
#pragma unroll
      for (int v = 0; v < 8; ++v) {
        int row = r * 16 + v + hi * 8;
        sh_proj[row * PS + c * 16 + m15] = f2bf_bits(fast_tanh(acc0[v] + acc1[v]));
      }
    }
  }
  __syncthreads();

  // ---- Step 2: GEMM2 wt[k][l] = sum_e proj[l][e] * cc[k][e] ----
  for (int t = w; t < 14; t += 8) {
    int r = t % 7, c2 = t / 7;
    const float* ccrow = cc + (size_t)(c2 * 16 + m15) * DIM + hi * 16;
    v16bf afr[8], bfr[8];
#pragma unroll
    for (int kk = 0; kk < 8; ++kk) {
      bfr[kk] = load_b16_g_f32(ccrow + kk * 32);
      afr[kk] = load_a_lds(sh_proj, PS, r * 16, kk * 32, lane);
    }
    v8f acc0 = {0.f, 0.f, 0.f, 0.f, 0.f, 0.f, 0.f, 0.f};
    v8f acc1 = {0.f, 0.f, 0.f, 0.f, 0.f, 0.f, 0.f, 0.f};
#pragma unroll
    for (int kk = 0; kk < 8; kk += 2) {
      acc0 = wmma_bf16(afr[kk],     bfr[kk],     acc0);
      acc1 = wmma_bf16(afr[kk + 1], bfr[kk + 1], acc1);
    }
#pragma unroll
    for (int v = 0; v < 8; ++v) {
      int row_l = r * 16 + v + hi * 8;
      sh_wt[(c2 * 16 + m15) * WTS + row_l] = acc0[v] + acc1[v];
    }
  }
  __syncthreads();

  // ---- Step 3: masked softmax over l per k; bf16 result, pad to 128 ----
  {
    const int nInt = num_int[b];
    for (int k = w; k < KC; k += 8) {
      if (k < nInt) {
        float mx = -3.4e38f;
        for (int l = lane; l < LQ; l += 32) mx = fmaxf(mx, sh_wt[k * WTS + l]);
#pragma unroll
        for (int off = 16; off > 0; off >>= 1) mx = fmaxf(mx, __shfl_xor(mx, off, 32));
        float s = 0.f;
        for (int l = lane; l < LQ; l += 32) s += __expf(sh_wt[k * WTS + l] - mx);
#pragma unroll
        for (int off = 16; off > 0; off >>= 1) s += __shfl_xor(s, off, 32);
        float inv = 1.0f / s;
        for (int l = lane; l < LP2; l += 32) {
          float v = (l < LQ) ? __expf(sh_wt[k * WTS + l] - mx) * inv : 0.f;
          sh_wtb[k * WBS + l] = f2bf_bits(v);
        }
      } else {
        // entire row masked to NEG -> softmax degenerates to uniform 1/L
        for (int l = lane; l < LP2; l += 32)
          sh_wtb[k * WBS + l] = f2bf_bits((l < LQ) ? (1.0f / LQ) : 0.f);
      }
    }
  }
  __syncthreads();

  // ---- Step 4: GEMM3 interests[k][d] = sum_l wt[k][l] * hist[l][d] ----
  for (int t = w; t < 32; t += 8) {
    int r3 = t >> 4, c3 = t & 15;
    v16bf afr[4], bfr[4];
#pragma unroll
    for (int kk = 0; kk < 4; ++kk) {
      afr[kk] = load_a_lds(sh_wtb, WBS, r3 * 16, kk * 32, lane);
      bfr[kk] = load_b16_lds_strided(sh_hist + (kk * 32 + hi * 16) * HS + c3 * 16 + m15, HS);
    }
    v8f acc0 = {0.f, 0.f, 0.f, 0.f, 0.f, 0.f, 0.f, 0.f};
    v8f acc1 = {0.f, 0.f, 0.f, 0.f, 0.f, 0.f, 0.f, 0.f};
#pragma unroll
    for (int kk = 0; kk < 4; kk += 2) {
      acc0 = wmma_bf16(afr[kk],     bfr[kk],     acc0);
      acc1 = wmma_bf16(afr[kk + 1], bfr[kk + 1], acc1);
    }
#pragma unroll
    for (int v = 0; v < 8; ++v) {
      int row_k = r3 * 16 + v + hi * 8;
      sh_int[row_k * IS + c3 * 16 + m15] = f2bf_bits(acc0[v] + acc1[v]);
    }
  }
  __syncthreads();

  // ---- Step 5: GEMM4 aw[n][k] = sum_d cand[n][d] * interests[k][d] ----
  if (w < 2) {
    int c4 = w;
    const float* crow = cand + ((size_t)b * NC + m15) * DIM + hi * 16;
    v16bf afr[8], bfr[8];
#pragma unroll
    for (int kk = 0; kk < 8; ++kk) {
      afr[kk] = load_a_g_f32(crow + kk * 32);
      bfr[kk] = load_b16_lds(sh_int + (c4 * 16 + m15) * IS + kk * 32 + hi * 16);
    }
    v8f acc0 = {0.f, 0.f, 0.f, 0.f, 0.f, 0.f, 0.f, 0.f};
    v8f acc1 = {0.f, 0.f, 0.f, 0.f, 0.f, 0.f, 0.f, 0.f};
#pragma unroll
    for (int kk = 0; kk < 8; kk += 2) {
      acc0 = wmma_bf16(afr[kk],     bfr[kk],     acc0);
      acc1 = wmma_bf16(afr[kk + 1], bfr[kk + 1], acc1);
    }
#pragma unroll
    for (int v = 0; v < 8; ++v) {
      int row_n = v + hi * 8;
      sh_aw[row_n * AWS + c4 * 16 + m15] = acc0[v] + acc1[v];
    }
  }
  __syncthreads();

  // ---- Step 6: stable descending top-dyn_K mask ----
  {
    int cat = cat_cnt[b];
    int x = 2 * cat;
    int dk = 32 - __clz(x - 1);              // ceil(log2(2*cat))
    dk = min(max(dk, 1), 32);
    for (int idx = tid; idx < NC * KC; idx += 256) {
      int n = idx >> 5, k = idx & 31;
      float v = sh_aw[n * AWS + k];
      int cnt = 0;
#pragma unroll
      for (int j = 0; j < KC; ++j) {
        float u = sh_aw[n * AWS + j];
        cnt += (u > v) || (u == v && j < k);  // rank via argsort(argsort) semantics
      }
      sh_awb[n * AWS + k] = f2bf_bits((cnt < dk) ? v : 0.0f);
    }
  }
  __syncthreads();

  // ---- Step 7: GEMM5 user[n][d] = sum_k masked[n][k] * interests[k][d] ----
  for (int c5 = w; c5 < 16; c5 += 8) {
    v8f acc = {0.f, 0.f, 0.f, 0.f, 0.f, 0.f, 0.f, 0.f};
    v16bf a  = load_a_lds(sh_awb, AWS, 0, 0, lane);
    v16bf bf = load_b16_lds_strided(sh_int + (hi * 16) * IS + c5 * 16 + m15, IS);
    acc = wmma_bf16(a, bf, acc);
#pragma unroll
    for (int v = 0; v < 8; ++v) {
      int row_n = v + hi * 8;
      out[((size_t)b * NC + row_n) * DIM + c5 * 16 + m15] = acc[v];
    }
  }
}

extern "C" void kernel_launch(void* const* d_in, const int* in_sizes, int n_in,
                              void* d_out, int out_size, void* d_ws, size_t ws_size,
                              hipStream_t stream) {
  (void)in_sizes; (void)n_in; (void)out_size; (void)d_ws; (void)ws_size;
  const float* hist    = (const float*)d_in[0];
  // d_in[1] = history_mask (unused by the reference computation)
  const float* cand    = (const float*)d_in[2];
  const int*   num_int = (const int*)d_in[3];
  const int*   cat_cnt = (const int*)d_in[4];
  const float* Wp      = (const float*)d_in[5];
  const float* cc      = (const float*)d_in[6];
  float*       out     = (float*)d_out;

  hipLaunchKernelGGL(poly_attn_kernel, dim3(BSZ), dim3(256), SMEM_BYTES, stream,
                     hist, cand, num_int, cat_cnt, Wp, cc, out);
}
